// SirenRadianceField_30520037605522
// MI455X (gfx1250) — matrix operand order repaired and checked by reading, not compiled, not run on hardware
//
#include <hip/hip_runtime.h>
#include <hip/hip_bf16.h>
#include <math.h>

typedef __attribute__((ext_vector_type(8)))  _Float16 v8h;
typedef __attribute__((ext_vector_type(16))) _Float16 v16h;
typedef __attribute__((ext_vector_type(8)))  float    v8f;

#define ZB 4
#define RR 2048
#define PP 64
#define HH 64
#define NHH 8
#define DD 2048
#define ZZ 256
#define PTS (ZB*RR*PP)            /* 524288 */
#define PTS_PER_BATCH (RR*PP)     /* 131072 */
#define TPB 256
#define WAVES 8
#define WG_PTS 128
#define STR 72                    /* padded LDS row stride in halves: 144B = 16B aligned, 36-dword bank shift */

// ---------------------------------------------------------------------------
// mapping network (tiny, f32 VALU)
// ---------------------------------------------------------------------------
__global__ void zn_kernel(const float* __restrict__ z, float* __restrict__ zn) {
  __shared__ float red[ZZ];
  int b = blockIdx.x, t = threadIdx.x;
  float v = z[b*ZZ + t];
  red[t] = v*v;
  __syncthreads();
  for (int s = ZZ/2; s > 0; s >>= 1) {
    if (t < s) red[t] += red[t + s];
    __syncthreads();
  }
  float nrm = fmaxf(sqrtf(red[0]), 1e-12f);
  zn[b*ZZ + t] = v / nrm;
}

template<int LEAKY>   // LEAKY=1: EqualLinear (0.1 scale) + LeakyReLU(0.2); 0: plain affine
__global__ void linear_kernel(const float* __restrict__ x, const float* __restrict__ W,
                              const float* __restrict__ bias, float* __restrict__ out,
                              int In, int Out) {
  int idx = blockIdx.x * blockDim.x + threadIdx.x;
  if (idx >= ZB * Out) return;
  int b = idx / Out, o = idx - b*Out;
  const float* xr = x + (size_t)b*In;
  const float* wr = W + (size_t)o*In;
  float a0 = 0.f, a1 = 0.f, a2 = 0.f, a3 = 0.f;
  for (int k = 0; k < In; k += 4) {     // In is a multiple of 4 (256/2048)
    a0 = fmaf(xr[k+0], wr[k+0], a0);
    a1 = fmaf(xr[k+1], wr[k+1], a1);
    a2 = fmaf(xr[k+2], wr[k+2], a2);
    a3 = fmaf(xr[k+3], wr[k+3], a3);
  }
  float acc = (a0+a1)+(a2+a3);
  if (LEAKY) {
    acc = 0.1f*acc + 0.1f*bias[o];
    acc = acc >= 0.f ? acc : 0.2f*acc;
  } else {
    acc += bias[o];
  }
  out[idx] = acc;
}

// pack the 9 trunk 64x64 matrices (swh[0..6], slw, rsw) into an f16 blob [9][64][64]
__global__ void pack_f16(const float* __restrict__ swh, const float* __restrict__ slw,
                         const float* __restrict__ rsw, _Float16* __restrict__ out) {
  int i = blockIdx.x * blockDim.x + threadIdx.x;
  if (i >= 9*64*64) return;
  float v;
  if (i < 7*4096)      v = swh[i];
  else if (i < 8*4096) v = slw[i - 7*4096];
  else                 v = rsw[i - 8*4096];
  out[i] = (_Float16)v;
}

// ---------------------------------------------------------------------------
// trunk: WMMA SIREN layers
// ---------------------------------------------------------------------------
__device__ __forceinline__ void wave_lds_fence() {
  // wave-private LDS tiles: only need this wave's DS ops drained (CDNA5 split counters)
  __builtin_amdgcn_wave_barrier();
  asm volatile("s_wait_dscnt 0" ::: "memory");
  __builtin_amdgcn_wave_barrier();
}

// A fragment 16x32 f16 (ISA 7.12.2): lane<16 row M=l16 holds K'=0..7 (v0-3), 16..23 (v4-7);
// lanes>=16 hold K'=8..15 / 24..31 of the same rows.
__device__ __forceinline__ v16h load_a_frag(const _Float16* row, int c, int lh) {
  v8h lo = *reinterpret_cast<const v8h*>(row + c*32 + lh*8);
  v8h hi = *reinterpret_cast<const v8h*>(row + c*32 + lh*8 + 16);
  v16h a;
  #pragma unroll
  for (int i = 0; i < 8; ++i) { a[i] = lo[i]; a[i+8] = hi[i]; }
  return a;
}

// B fragment 32x16 f16: lane<16 col N=l16 holds K'=0..7 (v0-3), 16..23 (v4-7); lanes>=16 hold K'=8..15 / 24..31.
__device__ __forceinline__ v16h load_b_frag(const _Float16* wl, int t, int c,
                                            int l16, int lh) {
  const _Float16* col = wl + (16*t + l16)*STR + c*32 + lh*8;
  v8h lo = *reinterpret_cast<const v8h*>(col);
  v8h hi = *reinterpret_cast<const v8h*>(col + 16);
  v16h b;
  #pragma unroll
  for (int i = 0; i < 8; ++i) { b[i] = lo[i]; b[i+8] = hi[i]; }
  return b;
}

template<bool FILM>
__device__ __forceinline__ void siren_layer(const _Float16* src,
                                            const _Float16* wl,
                                            _Float16* dst,
                                            const float* bias,
                                            const float* gam,
                                            const float* bet,
                                            int l16, int lh) {
  wave_lds_fence();                      // previous layer's stores -> this layer's loads
  const _Float16* arow = src + l16*STR;  // row M = l16 (points)
  v16h A0 = load_a_frag(arow, 0, lh);
  v16h A1 = load_a_frag(arow, 1, lh);
  #pragma unroll
  for (int t = 0; t < 4; ++t) {          // 4 output-neuron tiles of 16
    v8f C = {};
    v16h B0 = load_b_frag(wl, t, 0, l16, lh);
    C = __builtin_amdgcn_wmma_f32_16x16x32_f16(false, A0, false, B0, (short)0, C, false, false);
    v16h B1 = load_b_frag(wl, t, 1, l16, lh);
    C = __builtin_amdgcn_wmma_f32_16x16x32_f16(false, A1, false, B1, (short)0, C, false, false);
    int n = 16*t + l16;                  // D layout: N = lane%16 (+tile), M = vgpr + 8*(lane/16)
    float bi = bias[n];
    float g  = FILM ? gam[n] : 0.f;
    float bb = FILM ? bet[n] : 0.f;
    #pragma unroll
    for (int r = 0; r < 8; ++r) {
      int m = r + 8*lh;
      float v = C[r] + bi;
      if (FILM) v = fmaf(v, g, bb);
      dst[m*STR + n] = (_Float16)__sinf(v);   // w0 == 1 for all WMMA layers
    }
  }
}

__global__ __launch_bounds__(TPB)
void trunk_kernel(const float* __restrict__ origins, const float* __restrict__ directions,
                  const float* __restrict__ lengths,
                  const float* __restrict__ sw0, const float* __restrict__ sb0,
                  const _Float16* __restrict__ wf16,
                  const float* __restrict__ sbh, const float* __restrict__ slb,
                  const float* __restrict__ rsb,
                  const float* __restrict__ gam8, const float* __restrict__ bet8,
                  const float* __restrict__ rgamv, const float* __restrict__ rbetv,
                  const float* __restrict__ aw, const float* __restrict__ ab,
                  const float* __restrict__ cw, const float* __restrict__ cb,
                  float* __restrict__ out_alpha, float* __restrict__ out_rgb) {
  __shared__ __align__(16) _Float16 wlds[9*64*STR];           // 82944 B
  __shared__ __align__(16) _Float16 act[WAVES*2*16*STR];      // 36864 B
  __shared__ float sgam[8*64], sbet[8*64], srg[64], srbv[64];
  __shared__ float sbias[9*64], sw0p[64*4], sawv[64], scwv[3*64];

  const int tid  = threadIdx.x;
  const int base = blockIdx.x * WG_PTS;
  const int bat  = base / PTS_PER_BATCH;                      // whole WG in one batch

  // kick the weight blob toward our caches before the staging loop (global_prefetch_b8)
  __builtin_prefetch((const char*)wf16 + (size_t)tid * 288, 0, 3);

  // ---- stage weights (v8h = b128 copies) and per-batch FiLM params ----
  for (int idx = tid; idx < 9*64*8; idx += TPB) {
    int row = idx >> 3, kv = idx & 7;
    *reinterpret_cast<v8h*>(wlds + row*STR + kv*8) =
        *reinterpret_cast<const v8h*>(wf16 + row*64 + kv*8);
  }
  for (int idx = tid; idx < 512; idx += TPB) {
    sgam[idx] = gam8[bat*512 + idx];
    sbet[idx] = bet8[bat*512 + idx];
  }
  for (int idx = tid; idx < 9*64; idx += TPB)
    sbias[idx] = (idx < 448) ? sbh[idx] : ((idx < 512) ? slb[idx-448] : rsb[idx-512]);
  if (tid < 64) {
    srg[tid]  = rgamv[bat*64 + tid];
    srbv[tid] = rbetv[bat*64 + tid];
    sw0p[tid*4+0] = sw0[tid*3+0];
    sw0p[tid*4+1] = sw0[tid*3+1];
    sw0p[tid*4+2] = sw0[tid*3+2];
    sw0p[tid*4+3] = sb0[tid];
    sawv[tid] = aw[tid];
  }
  if (tid < 192) scwv[tid] = cw[tid];
  __syncthreads();

  const int wave = tid >> 5, lane = tid & 31;
  const int l16 = lane & 15, lh = lane >> 4;
  _Float16* t0 = act + wave*(2*16*STR);      // direct expressions over __shared__:
  _Float16* t1 = t0 + 16*STR;                // keeps LDS address-space inference intact
  const int tbase = base + wave*16;

  // ---- first SIREN layer (3 -> 64, w0 = 30, FiLM head 0): pure VALU, vector stores ----
  {
    int pid = tbase + l16;
    int rp  = pid - bat*PTS_PER_BATCH;
    int r   = rp >> 6;                                // / P
    float len = lengths[pid];
    size_t oix = ((size_t)bat*RR + r)*3;
    float px = fmaf(directions[oix+0], len, origins[oix+0]);
    float py = fmaf(directions[oix+1], len, origins[oix+1]);
    float pz = fmaf(directions[oix+2], len, origins[oix+2]);
    #pragma unroll
    for (int jv = 0; jv < 4; ++jv) {
      v8h pack;
      #pragma unroll
      for (int i = 0; i < 8; ++i) {
        int n = lh*32 + jv*8 + i;
        float a = fmaf(sw0p[n*4+2], pz, fmaf(sw0p[n*4+1], py, fmaf(sw0p[n*4+0], px, sw0p[n*4+3])));
        a = fmaf(a, sgam[n], sbet[n]);
        pack[i] = (_Float16)__sinf(30.f * a);
      }
      *reinterpret_cast<v8h*>(t0 + l16*STR + lh*32 + jv*8) = pack;
    }
  }

  // ---- 7 FiLM'd hidden layers (64x64 WMMA); compile-time ping-pong via full unroll ----
  #pragma unroll
  for (int L = 0; L < 7; ++L) {
    if ((L & 1) == 0)
      siren_layer<true>(t0, wlds + L*64*STR, t1,
                        sbias + L*64, sgam + (L+1)*64, sbet + (L+1)*64, l16, lh);
    else
      siren_layer<true>(t1, wlds + L*64*STR, t0,
                        sbias + L*64, sgam + (L+1)*64, sbet + (L+1)*64, l16, lh);
  }
  // after L=6 (even) activations are in t1
  // ---- last SIREN layer (no FiLM): x -> t0 ----
  siren_layer<false>(t1, wlds + 7*64*STR, t0, sbias + 7*64, nullptr, nullptr, l16, lh);
  // ---- rs layer (FiLM rgam/rbet): xr -> t1 ----
  siren_layer<true>(t0, wlds + 8*64*STR, t1, sbias + 8*64, srg, srbv, l16, lh);

  wave_lds_fence();

  // ---- heads: 16 points x (alpha + rgb[3]) = 64 dots, 2 per lane, v8h reads ----
  float ab0 = ab[0];
  float c0 = cb[0], c1 = cb[1], c2 = cb[2];
  #pragma unroll
  for (int q = 0; q < 2; ++q) {
    int d = lane*2 + q;
    int m = d >> 2, ch = d & 3;
    const _Float16* xt = (ch == 0 ? t0 : t1) + m*STR;      // x in t0, xr in t1
    const float* wv = (ch == 0) ? sawv : (scwv + (ch-1)*64);
    float acc = (ch == 0) ? ab0 : (ch == 1 ? c0 : (ch == 2 ? c1 : c2));
    #pragma unroll
    for (int kv = 0; kv < 8; ++kv) {
      v8h hv = *reinterpret_cast<const v8h*>(xt + kv*8);
      #pragma unroll
      for (int i = 0; i < 8; ++i) acc = fmaf((float)hv[i], wv[kv*8 + i], acc);
    }
    float sg = 1.f / (1.f + __expf(-acc));
    int pid = tbase + m;
    if (ch == 0) out_alpha[pid] = sg;
    else         out_rgb[(size_t)pid*3 + (ch-1)] = sg;
  }
}

// ---------------------------------------------------------------------------
extern "C" void kernel_launch(void* const* d_in, const int* in_sizes, int n_in,
                              void* d_out, int out_size, void* d_ws, size_t ws_size,
                              hipStream_t stream) {
  const float* z    = (const float*)d_in[0];
  const float* orig = (const float*)d_in[1];
  const float* dirs = (const float*)d_in[2];
  const float* lens = (const float*)d_in[3];
  const float* mw0 = (const float*)d_in[4];  const float* mb0 = (const float*)d_in[5];
  const float* mw1 = (const float*)d_in[6];  const float* mb1 = (const float*)d_in[7];
  const float* mw2 = (const float*)d_in[8];  const float* mb2 = (const float*)d_in[9];
  const float* mgw = (const float*)d_in[10]; const float* mgb = (const float*)d_in[11];
  const float* mbw = (const float*)d_in[12]; const float* mbb = (const float*)d_in[13];
  const float* rw0 = (const float*)d_in[14]; const float* rb0 = (const float*)d_in[15];
  const float* rw1 = (const float*)d_in[16]; const float* rb1 = (const float*)d_in[17];
  const float* rw2 = (const float*)d_in[18]; const float* rb2 = (const float*)d_in[19];
  const float* rgw = (const float*)d_in[20]; const float* rgbb= (const float*)d_in[21];
  const float* rbw = (const float*)d_in[22]; const float* rbb = (const float*)d_in[23];
  const float* sw0 = (const float*)d_in[24]; const float* sb0 = (const float*)d_in[25];
  const float* swh = (const float*)d_in[26]; const float* sbh = (const float*)d_in[27];
  const float* slw = (const float*)d_in[28]; const float* slb = (const float*)d_in[29];
  const float* aw  = (const float*)d_in[30]; const float* ab  = (const float*)d_in[31];
  const float* rsw = (const float*)d_in[32]; const float* rsb = (const float*)d_in[33];
  const float* cw  = (const float*)d_in[34]; const float* cb  = (const float*)d_in[35];

  float* ws    = (float*)d_ws;
  float* zn    = ws;                    // 4*256
  float* mx0   = zn    + ZB*ZZ;         // 4*2048
  float* mx1   = mx0   + ZB*DD;         // 4*2048
  float* gamma = mx1   + ZB*DD;         // 4*512
  float* beta  = gamma + ZB*NHH*HH;     // 4*512
  float* rx0   = beta  + ZB*NHH*HH;     // 4*256
  float* rx1   = rx0   + ZB*ZZ;         // 4*256
  float* rgamv = rx1   + ZB*ZZ;         // 4*64
  float* rbetv = rgamv + ZB*HH;         // 4*64
  _Float16* wf16 = (_Float16*)(rbetv + ZB*HH);  // 9*4096 halves (16B-aligned offset)

  auto cdiv = [](int a, int b){ return (a + b - 1) / b; };
  const int th = 256;

  zn_kernel<<<ZB, ZZ, 0, stream>>>(z, zn);
  // style mapping: 3 EqualLinear layers + gamma/beta heads
  linear_kernel<1><<<cdiv(ZB*DD, th),  th, 0, stream>>>(zn,  mw0, mb0, mx0, ZZ, DD);
  linear_kernel<1><<<cdiv(ZB*DD, th),  th, 0, stream>>>(mx0, mw1, mb1, mx1, DD, DD);
  linear_kernel<1><<<cdiv(ZB*DD, th),  th, 0, stream>>>(mx1, mw2, mb2, mx0, DD, DD);
  linear_kernel<0><<<cdiv(ZB*512, th), th, 0, stream>>>(mx0, mgw, mgb, gamma, DD, NHH*HH);
  linear_kernel<0><<<cdiv(ZB*512, th), th, 0, stream>>>(mx0, mbw, mbb, beta,  DD, NHH*HH);
  // ray mapping
  linear_kernel<1><<<cdiv(ZB*ZZ, th),  th, 0, stream>>>(zn,  rw0, rb0, rx0, ZZ, ZZ);
  linear_kernel<1><<<cdiv(ZB*ZZ, th),  th, 0, stream>>>(rx0, rw1, rb1, rx1, ZZ, ZZ);
  linear_kernel<1><<<cdiv(ZB*ZZ, th),  th, 0, stream>>>(rx1, rw2, rb2, rx0, ZZ, ZZ);
  linear_kernel<0><<<cdiv(ZB*HH, th),  th, 0, stream>>>(rx0, rgw, rgbb, rgamv, ZZ, HH);
  linear_kernel<0><<<cdiv(ZB*HH, th),  th, 0, stream>>>(rx0, rbw, rbb,  rbetv, ZZ, HH);
  // f16 trunk weights
  pack_f16<<<cdiv(9*64*64, th), th, 0, stream>>>(swh, slw, rsw, wf16);

  float* out_alpha = (float*)d_out;
  float* out_rgb   = out_alpha + PTS;
  trunk_kernel<<<PTS/WG_PTS, TPB, 0, stream>>>(orig, dirs, lens, sw0, sb0, wf16,
      sbh, slb, rsb, gamma, beta, rgamv, rbetv, aw, ab, cw, cb, out_alpha, out_rgb);
}
